// HashedLinear_60584808678023
// MI455X (gfx1250) — matrix operand order repaired and checked by reading, not compiled, not run on hardware
//
#include <hip/hip_runtime.h>
#include <hip/hip_bf16.h>
#include <stdint.h>

typedef __attribute__((ext_vector_type(16))) __bf16        v16bf;
typedef __attribute__((ext_vector_type(8)))  float         v8f;
typedef __attribute__((ext_vector_type(4)))  unsigned int  v4u;

#define GLOBAL_AS __attribute__((address_space(1)))
typedef GLOBAL_AS const unsigned short* gcu16p;
typedef GLOBAL_AS const v4u*            gcv4p;
typedef GLOBAL_AS const float*          gcf32p;
typedef GLOBAL_AS float*                gf32p;

__device__ __forceinline__ unsigned short f2bf_rne(float f) {
  unsigned int u = __float_as_uint(f);
  u += 0x7fffu + ((u >> 16) & 1u);   // round-to-nearest-even
  return (unsigned short)(u >> 16);
}

// ---------------- x / x_hash  f32 -> bf16 (4 elems / thread) ----------------
__global__ void cvt_f32_bf16(const float* __restrict__ src,
                             unsigned short* __restrict__ dst, int n4) {
  int i = blockIdx.x * blockDim.x + threadIdx.x;
  if (i >= n4) return;
  float4 v = ((const float4*)src)[i];
  uint2 o;
  o.x = (unsigned)f2bf_rne(v.x) | ((unsigned)f2bf_rne(v.y) << 16);
  o.y = (unsigned)f2bf_rne(v.z) | ((unsigned)f2bf_rne(v.w) << 16);
  ((uint2*)dst)[i] = o;
}

// ---- fused: W -> bf16  AND  hash_weight = softmax-bucket transform of W ----
__global__ void prep_weights(const float* __restrict__ W,
                             const float* __restrict__ a0,
                             const float* __restrict__ a1,
                             const float* __restrict__ a2,
                             const float* __restrict__ values,
                             unsigned short* __restrict__ wbf,
                             unsigned short* __restrict__ hwbf,
                             int total, float T) {
  __shared__ float s0[16], s1[16], s2[16], sv[16];
  if (threadIdx.x < 16) {
    s0[threadIdx.x] = a0[threadIdx.x];
    s1[threadIdx.x] = a1[threadIdx.x];
    s2[threadIdx.x] = a2[threadIdx.x];
    sv[threadIdx.x] = values[threadIdx.x];
  }
  __syncthreads();
  int i = blockIdx.x * blockDim.x + threadIdx.x;
  if (i >= total) return;
  float w  = W[i];
  float w2 = w * w;
  float fx[16];
  float mx = -3.4e38f;
#pragma unroll
  for (int b = 0; b < 16; ++b) {
    float f = T * (s1[b] * w + s2[b] * w2 + s0[b]);
    fx[b] = f;
    mx = fmaxf(mx, f);
  }
  float den = 0.f, num = 0.f;
#pragma unroll
  for (int b = 0; b < 16; ++b) {
    float e = __expf(fx[b] - mx);
    den += e;
    num += e * sv[b];
  }
  wbf[i]  = f2bf_rne(w);
  hwbf[i] = f2bf_rne(num / den);
}

// ------------------------- bf16 WMMA GEMM -----------------------------------
// out[r, c] = sum_k A[r, k] * Wt[c, k] + bias[c]
// A: [Bdim x Mdim] bf16 row-major (K contiguous); Wt: [Ndim x Mdim] bf16.
// Block = 256 thr = 8 waves arranged 2(rows) x 4(cols);
// wave tile 64 x 64 = 4x4 WMMA accumulators; block tile 128 x 256; K step 32.
// All inner-loop memory ops pinned to address_space(1) => global_load/store.
__global__ __launch_bounds__(256)
void gemm_bf16(const unsigned short* __restrict__ A,
               const unsigned short* __restrict__ Wt,
               const float* __restrict__ bias,
               float* __restrict__ out,
               int Bdim, int Ndim, int Mdim) {
  const int lane = threadIdx.x & 31;
  const int wave = threadIdx.x >> 5;
  const int ln   = lane & 15;        // lane-in-half
  const int lh   = lane >> 4;        // half-wave select

  const int rowBase = blockIdx.x * 128 + (wave & 1) * 64;   // output rows
  const int colBase = blockIdx.y * 256 + (wave >> 1) * 64;  // output cols

  v8f acc[4][4];
#pragma unroll
  for (int mi = 0; mi < 4; ++mi)
#pragma unroll
    for (int ni = 0; ni < 4; ++ni) acc[mi][ni] = (v8f){0,0,0,0,0,0,0,0};

  union U { v4u u[2]; v16bf v; };

  // Per-row global pointers, advanced by 32 elements (64 B) per K step.
  // A tile (ISA 16-bit A 16x32 layout): lane ln holds M=ln,
  //   K = lh*8 + {0..7}  and  16 + lh*8 + {0..7}  -> chunks at +0 and +2 v4u.
  // B tile (32x16): lane holds column n=ln, K = lh*8 + {0..7} and 16 + lh*8 + {0..7} -> chunks at +0 and +2 v4u.
  gcu16p ap[4];
  gcu16p bp[4];
#pragma unroll
  for (int mi = 0; mi < 4; ++mi)
    ap[mi] = (gcu16p)A + (size_t)(rowBase + mi * 16 + ln) * Mdim + lh * 8;
#pragma unroll
  for (int ni = 0; ni < 4; ++ni)
    bp[ni] = (gcu16p)Wt + (size_t)(colBase + ni * 16 + ln) * Mdim + lh * 8;

  for (int kk = 0; kk < Mdim; kk += 32) {
    v16bf a[4], b[4];
#pragma unroll
    for (int mi = 0; mi < 4; ++mi) {
      gcv4p p = (gcv4p)ap[mi];
      U ua; ua.u[0] = p[0]; ua.u[1] = p[2];   // +16 elements = +2 v4u
      a[mi] = ua.v;
      ap[mi] += 32;
    }
#pragma unroll
    for (int ni = 0; ni < 4; ++ni) {
      gcv4p q = (gcv4p)bp[ni];
      U ub; ub.u[0] = q[0]; ub.u[1] = q[2];
      b[ni] = ub.v;
      bp[ni] += 32;
    }
#pragma unroll
    for (int mi = 0; mi < 4; ++mi)
#pragma unroll
      for (int ni = 0; ni < 4; ++ni)
        acc[mi][ni] = __builtin_amdgcn_wmma_f32_16x16x32_bf16(
            false, a[mi], false, b[ni], (short)0, acc[mi][ni], false, false);
  }

  // D layout: VGPR v -> row (v + lh*8), col ln.
  gf32p  outg  = (gf32p)out;
  gcf32p biasg = (gcf32p)bias;
#pragma unroll
  for (int ni = 0; ni < 4; ++ni) {
    const int col = colBase + ni * 16 + ln;
    const float bv = biasg[col];
#pragma unroll
    for (int mi = 0; mi < 4; ++mi) {
      const int row0 = rowBase + mi * 16 + lh * 8;
#pragma unroll
      for (int v = 0; v < 8; ++v)
        outg[(size_t)(row0 + v) * Ndim + col] = acc[mi][ni][v] + bv;
    }
  }
}

extern "C" void kernel_launch(void* const* d_in, const int* in_sizes, int n_in,
                              void* d_out, int out_size, void* d_ws, size_t ws_size,
                              hipStream_t stream) {
  const float* x      = (const float*)d_in[0];
  const float* x_hash = (const float*)d_in[1];
  const float* W      = (const float*)d_in[2];
  const float* bias   = (const float*)d_in[3];
  const float* a0     = (const float*)d_in[4];
  const float* a1     = (const float*)d_in[5];
  const float* a2     = (const float*)d_in[6];
  const float* values = (const float*)d_in[7];

  const int NM   = in_sizes[2];          // N*M
  const int N    = in_sizes[3];          // bias length
  const int M    = NM / N;
  const int Bdim = in_sizes[0] / M;

  // Workspace layout (all bf16/ushort): xbf | xhbf | wbf | hwbf  (~50 MB total)
  unsigned short* xbf  = (unsigned short*)d_ws;
  unsigned short* xhbf = xbf  + (size_t)Bdim * M;
  unsigned short* wbf  = xhbf + (size_t)Bdim * M;
  unsigned short* hwbf = wbf  + (size_t)NM;

  float* out  = (float*)d_out;
  float* outh = out + (size_t)Bdim * N;

  const int n4 = (Bdim * M) / 4;
  cvt_f32_bf16<<<(n4 + 255) / 256, 256, 0, stream>>>(x, xbf, n4);
  cvt_f32_bf16<<<(n4 + 255) / 256, 256, 0, stream>>>(x_hash, xhbf, n4);
  prep_weights<<<(NM + 255) / 256, 256, 0, stream>>>(W, a0, a1, a2, values,
                                                     wbf, hwbf, NM, 100.0f);

  dim3 grid(Bdim / 128, N / 256);        // 32 x 8 blocks for 4096x2048
  gemm_bf16<<<grid, 256, 0, stream>>>(xbf,  wbf,  bias, out,  Bdim, N, M);
  gemm_bf16<<<grid, 256, 0, stream>>>(xhbf, hwbf, bias, outh, Bdim, N, M);
}
